// SlotCrossAttention_30004641529994
// MI455X (gfx1250) — matrix operand order repaired and checked by reading, not compiled, not run on hardware
//
#include <hip/hip_runtime.h>
#include <hip/hip_bf16.h>

// ---------------------------------------------------------------------------
// SlotCrossAttention for MI455X (gfx1250, wave32, WMMA bf16).
// B=8, Q=100 (padded 112), C=1024, K=4096 keys, NH=16, HD=64, scale=1/8.
//
// Pipeline (all bf16 WMMA, fp32 accumulate):
//   1) Kp  = bf16((mem+mpos) @ Wk.T)   stored [B,NH,4096,64]
//   2) Vt  = bf16( mem       @ Wv.T)   stored [B,NH,64,4096] (transposed)
//   3) Qp  = bf16(((tgt+qpos) @ Wq.T) * scale) stored [B,NH,112,64]
//   4) attention per (b,h): stream 128-key tiles; V tile staged in LDS
//   5) preLN = updates @ Wo.T + bo + tgt (WMMA)
//   6) LayerNorm rows -> d_out
// Weight tiles staged through LDS (bf16) so the 8 waves of a block share one
// global fetch + one fp32->bf16 conversion instead of eight.
// ---------------------------------------------------------------------------

typedef __bf16 bf16;
typedef __attribute__((ext_vector_type(16))) __bf16 v16bf;
typedef __attribute__((ext_vector_type(8)))  __bf16 v8bf;
typedef __attribute__((ext_vector_type(8)))  float  v8f;

__device__ __forceinline__ v8f wmma_bf16(v16bf a, v16bf b, v8f c) {
  // D = A(16x32 bf16) * B(32x16 bf16) + C(16x16 f32)
  return __builtin_amdgcn_wmma_f32_16x16x32_bf16(false, a, false, b,
                                                 (short)0, c, false, false);
}

#define BATCH 8
#define QLEN  100
#define QPAD  112
#define CDIM  1024
#define NHEAD 16
#define HDIM  64
#define KEYS  4096
#define QKSCALE 0.125f

#define WSTRIDE 272   // bf16 elements per LDS weight row (64 rows x 256 K + pad)
#define VSTRIDE 136   // bf16 elements per LDS V row (64 rows x 128 keys + pad)

// ---------------------------------------------------------------------------
// Projection GEMM: out[m,n] = sum_k A[m,k] * W[n,k]   (torch Linear)
// Block: 256 threads (8 waves). Block tile: 128(M) x 64(N).
// Wave w owns M-rows [w*16, w*16+16), all 64 N cols (4 WMMA accumulators).
// Weight chunk (64 N x 256 K) cooperatively staged fp32->bf16 into LDS once
// per block; B-fragments then come from LDS (shared by all 8 waves).
// MODE 0: store Kp [B,NH,4096,64]
// MODE 1: store Vt [B,NH,64,4096]  (transposed; packed 16B bf16 stores)
// MODE 2: store Qp [B,NH,112,64], A rows padded (q>=100 -> 0), *QKSCALE
// MODE 3: A is bf16 (updates), out fp32 preLN = acc + bo[n] + tgt[b,q,n]
// ---------------------------------------------------------------------------
template <int MODE>
__global__ __launch_bounds__(256) void gemm_kernel(
    const float* __restrict__ A, const float* __restrict__ A2,
    const bf16* __restrict__ Abf, const float* __restrict__ W,
    const float* __restrict__ bias, const float* __restrict__ resid,
    bf16* __restrict__ outb, float* __restrict__ outf) {
  __shared__ bf16 Wl[64 * WSTRIDE];  // 34.8 KB

  const int wv   = threadIdx.x >> 5;
  const int lane = threadIdx.x & 31;
  const int l16  = lane & 15;
  const int hi   = lane >> 4;

  const int mrow  = blockIdx.x * 128 + wv * 16 + l16;  // padded global row
  const int nbase = blockIdx.y * 64;

  // Resolve A source row (per-lane; loads only, EXEC reconverges before WMMA).
  long arow = mrow;
  bool avalid = true;
  if (MODE == 2) {
    int b = mrow / QPAD, q = mrow % QPAD;
    arow = (long)b * QLEN + q;
    avalid = (q < QLEN);
  }

  v8f acc[4] = {};

  for (int kc = 0; kc < CDIM; kc += 256) {
    // ---- Cooperative weight stage: rows nbase..+64, K = kc..kc+256.
    {
      const int row = threadIdx.x >> 2;
      const int cb  = (threadIdx.x & 3) * 64;
      const float* wp = W + (long)(nbase + row) * CDIM + kc + cb;
      bf16* dst = Wl + row * WSTRIDE + cb;
#pragma unroll
      for (int u = 0; u < 4; u++) {
        float t4[16];
        *(float4*)(t4 + 0)  = *(const float4*)(wp + u * 16 + 0);
        *(float4*)(t4 + 4)  = *(const float4*)(wp + u * 16 + 4);
        *(float4*)(t4 + 8)  = *(const float4*)(wp + u * 16 + 8);
        *(float4*)(t4 + 12) = *(const float4*)(wp + u * 16 + 12);
        bf16 pk[16];
#pragma unroll
        for (int e = 0; e < 16; e++) pk[e] = (bf16)t4[e];
        *(v8bf*)(dst + u * 16 + 0) = *(v8bf*)(pk + 0);
        *(v8bf*)(dst + u * 16 + 8) = *(v8bf*)(pk + 8);
      }
    }
    __syncthreads();

    for (int k0 = kc; k0 < kc + 256; k0 += 32) {
      // ---- A fragment: lane row mrow, K-chunks {k0+8*hi..+8, k0+16+8*hi..+8}
      v16bf af;
      if (MODE == 3) {
        const bf16* ap = Abf + (long)mrow * CDIM + k0 + 8 * hi;
        v8bf c0 = *(const v8bf*)ap;
        v8bf c1 = *(const v8bf*)(ap + 16);
#pragma unroll
        for (int e = 0; e < 8; e++) { af[e] = c0[e]; af[8 + e] = c1[e]; }
      } else {
        float atmp[16];
        if (MODE == 2 && !avalid) {
#pragma unroll
          for (int e = 0; e < 16; e++) atmp[e] = 0.f;
        } else {
          const float* ap = A + arow * CDIM + k0 + 8 * hi;
          *(float4*)(atmp + 0)  = *(const float4*)(ap + 0);
          *(float4*)(atmp + 4)  = *(const float4*)(ap + 4);
          *(float4*)(atmp + 8)  = *(const float4*)(ap + 16);
          *(float4*)(atmp + 12) = *(const float4*)(ap + 20);
          if (A2) {
            const float* ap2 = A2 + arow * CDIM + k0 + 8 * hi;
            float4 b0 = *(const float4*)(ap2 + 0);
            float4 b1 = *(const float4*)(ap2 + 4);
            float4 b2 = *(const float4*)(ap2 + 16);
            float4 b3 = *(const float4*)(ap2 + 20);
            atmp[0] += b0.x; atmp[1] += b0.y; atmp[2]  += b0.z; atmp[3]  += b0.w;
            atmp[4] += b1.x; atmp[5] += b1.y; atmp[6]  += b1.z; atmp[7]  += b1.w;
            atmp[8] += b2.x; atmp[9] += b2.y; atmp[10] += b2.z; atmp[11] += b2.w;
            atmp[12]+= b3.x; atmp[13]+= b3.y; atmp[14] += b3.z; atmp[15] += b3.w;
          }
        }
#pragma unroll
        for (int e = 0; e < 16; e++) af[e] = (bf16)atmp[e];
      }

      // ---- B fragments from LDS: lane row n, K-chunks {(k0-kc)+8*hi..+8, (k0-kc)+16+8*hi..+8}
#pragma unroll
      for (int nt = 0; nt < 4; nt++) {
        const bf16* bp = Wl + (nt * 16 + l16) * WSTRIDE + (k0 - kc) + 8 * hi;
        v8bf c0 = *(const v8bf*)(bp + 0);
        v8bf c1 = *(const v8bf*)(bp + 16);
        v16bf bw;
#pragma unroll
        for (int e = 0; e < 8; e++) { bw[e] = c0[e]; bw[8 + e] = c1[e]; }
        acc[nt] = wmma_bf16(af, bw, acc[nt]);
      }
    }
    __syncthreads();
  }

  // ---- Stores. C fragment: lane col n = nbase+nt*16+l16, rows m = mbase+r.
  const int mbase = blockIdx.x * 128 + wv * 16 + hi * 8;
#pragma unroll
  for (int nt = 0; nt < 4; nt++) {
    const int n = nbase + nt * 16 + l16;
    if (MODE == 0) {  // Kp [B,NH,4096,64]
      const int h = n >> 6, d = n & 63;
#pragma unroll
      for (int r = 0; r < 8; r++) {
        const int m = mbase + r, b = m >> 12, key = m & 4095;
        outb[((((long)b * NHEAD + h) * KEYS + key) * HDIM) + d] = (bf16)acc[nt][r];
      }
    } else if (MODE == 1) {  // Vt [B,NH,64,4096], 8 consecutive keys per lane
      const int h = n >> 6, d = n & 63;
      const int b = mbase >> 12, key0 = mbase & 4095;
      bf16 pk[8];
#pragma unroll
      for (int r = 0; r < 8; r++) pk[r] = (bf16)acc[nt][r];
      *(v8bf*)(outb + (((long)b * NHEAD + h) * HDIM + d) * KEYS + key0) = *(v8bf*)pk;
    } else if (MODE == 2) {  // Qp [B,NH,112,64], scaled
      const int h = n >> 6, d = n & 63;
#pragma unroll
      for (int r = 0; r < 8; r++) {
        const int m = mbase + r, b = m / QPAD, q = m % QPAD;
        outb[(((long)b * NHEAD + h) * QPAD + q) * HDIM + d] =
            (bf16)(acc[nt][r] * QKSCALE);
      }
    } else {  // MODE 3: preLN fp32 [B,112,1024]
#pragma unroll
      for (int r = 0; r < 8; r++) {
        const int m = mbase + r, b = m / QPAD, q = m % QPAD;
        float v = acc[nt][r] + bias[n];
        if (q < QLEN) v += resid[((long)b * QLEN + q) * CDIM + n];
        outf[(long)m * CDIM + n] = v;
      }
    }
  }
}

// ---------------------------------------------------------------------------
// Attention kernel: one block per (b,h). 256 threads = 8 waves.
// Streams 32 tiles of 128 keys. LDS: S[112][132] f32, attn[112][128] bf16,
// rowsum[112] f32, Vl[64][136] bf16 (~103 KB dynamic LDS, <320 KB WGP).
// ---------------------------------------------------------------------------
__global__ __launch_bounds__(256) void attn_kernel(
    const bf16* __restrict__ Qp, const bf16* __restrict__ Kp,
    const bf16* __restrict__ Vt, bf16* __restrict__ upd) {
  extern __shared__ char smem[];
  float* S      = (float*)smem;                                   // [112][132]
  bf16*  Ab     = (bf16*)(smem + QPAD * 132 * 4);                 // [112][128]
  float* rowsum = (float*)(smem + QPAD * 132 * 4 + QPAD * 128 * 2);
  bf16*  Vl     = (bf16*)(smem + QPAD * 132 * 4 + QPAD * 128 * 2 + QPAD * 4);

  const int bh = blockIdx.x;
  const int b = bh >> 4, h = bh & 15;
  const int wv = threadIdx.x >> 5, lane = threadIdx.x & 31;
  const int l16 = lane & 15, hi = lane >> 4;

  const long qbase = (long)bh * QPAD * HDIM;
  const long kbase = (long)bh * KEYS * HDIM;
  const long vbase = (long)bh * HDIM * KEYS;

  // Preload Q A-fragments (scale already folded in): afq[mtile][kslice]
  v16bf afq[7][2];
#pragma unroll
  for (int mt = 0; mt < 7; mt++)
#pragma unroll
    for (int s = 0; s < 2; s++) {
      const bf16* p = Qp + qbase + (long)(mt * 16 + l16) * HDIM + s * 32 + 16 * hi;
      v8bf c0 = *(const v8bf*)p;
      v8bf c1 = *(const v8bf*)(p + 8);
#pragma unroll
      for (int e = 0; e < 8; e++) { afq[mt][s][e] = c0[e]; afq[mt][s][8 + e] = c1[e]; }
    }

  if (threadIdx.x < QPAD) rowsum[threadIdx.x] = 0.f;
  v8f U[4] = {};
  __syncthreads();

  for (int kt = 0; kt < 32; kt++) {
    const int kb = kt * 128;

    // ---- MMA1: S[q, key] tiles. Wave w owns key columns [w*16, w*16+16).
    {
      const int keyc = kb + wv * 16 + l16;
      v16bf bk[2];
#pragma unroll
      for (int s = 0; s < 2; s++)
        bk[s] = *(const v16bf*)(Kp + kbase + (long)keyc * HDIM + s * 32 + 16 * hi);
      const int j = wv * 16 + l16;
#pragma unroll
      for (int mt = 0; mt < 7; mt++) {
        v8f z = {};
        v8f a = wmma_bf16(afq[mt][0], bk[0], z);
        a = wmma_bf16(afq[mt][1], bk[1], a);
#pragma unroll
        for (int r = 0; r < 8; r++) {
          const int i = mt * 16 + hi * 8 + r;
          S[i * 132 + j] = (i < QLEN) ? a[r] : -1e30f;  // mask padded queries
        }
      }
    }

    // ---- Cooperative V-tile stage: Vl[d][key-rel], 64 x 128 bf16.
    {
      const int d  = threadIdx.x >> 2;
      const int cb = (threadIdx.x & 3) * 32;
      const bf16* vp = Vt + vbase + (long)d * KEYS + kb + cb;
      bf16* dst = Vl + d * VSTRIDE + cb;
      v8bf x0 = *(const v8bf*)(vp + 0);
      v8bf x1 = *(const v8bf*)(vp + 8);
      v8bf x2 = *(const v8bf*)(vp + 16);
      v8bf x3 = *(const v8bf*)(vp + 24);
      *(v8bf*)(dst + 0)  = x0;
      *(v8bf*)(dst + 8)  = x1;
      *(v8bf*)(dst + 16) = x2;
      *(v8bf*)(dst + 24) = x3;
    }
    __syncthreads();

    // ---- Column softmax over the 100 queries (threads 0..127 own a column).
    if (threadIdx.x < 128) {
      const int j = threadIdx.x;
      float m1 = -1e30f;
      for (int i = 0; i < QLEN; i++) m1 = fmaxf(m1, S[i * 132 + j]);
      float sum = 0.f;
      for (int i = 0; i < QLEN; i++) {
        float e = __expf(S[i * 132 + j] - m1);
        S[i * 132 + j] = e;
        sum += e;
      }
      const float inv = 1.f / sum;
      for (int i = 0; i < QLEN; i++) Ab[i * 128 + j] = (bf16)(S[i * 132 + j] * inv);
      for (int i = QLEN; i < QPAD; i++) Ab[i * 128 + j] = (bf16)0.f;
    }
    __syncthreads();

    // ---- Row sums (for key-axis renormalization), thread i owns query row i.
    if (threadIdx.x < QLEN) {
      const int i = threadIdx.x;
      float rs = 0.f;
      for (int j = 0; j < 128; j++) rs += (float)Ab[i * 128 + j];
      rowsum[i] += rs;
    }

    // ---- MMA2: U += attn(112x128) @ V(128x64). Wave w<7 owns query tile w.
    if (wv < 7) {
#pragma unroll
      for (int s = 0; s < 4; s++) {
        v16bf aa;
        const bf16* p = Ab + (wv * 16 + l16) * 128 + s * 32 + 16 * hi;
        v8bf c0 = *(const v8bf*)p;
        v8bf c1 = *(const v8bf*)(p + 8);
#pragma unroll
        for (int e = 0; e < 8; e++) { aa[e] = c0[e]; aa[8 + e] = c1[e]; }
#pragma unroll
        for (int nt = 0; nt < 4; nt++) {
          const bf16* vp = Vl + (nt * 16 + l16) * VSTRIDE + s * 32 + 16 * hi;
          v8bf d0 = *(const v8bf*)(vp + 0);
          v8bf d1 = *(const v8bf*)(vp + 8);
          v16bf bv;
#pragma unroll
          for (int e = 0; e < 8; e++) { bv[e] = d0[e]; bv[8 + e] = d1[e]; }
          U[nt] = wmma_bf16(aa, bv, U[nt]);
        }
      }
    }
    __syncthreads();
  }

  // ---- updates = U / (rowsum + 1e-8) -> upd [B,112,1024] bf16
  if (wv < 7) {
#pragma unroll
    for (int nt = 0; nt < 4; nt++) {
      const int d = nt * 16 + l16;
#pragma unroll
      for (int r = 0; r < 8; r++) {
        const int q = wv * 16 + hi * 8 + r;
        const float rs = rowsum[q] + 1e-8f;
        upd[((long)b * QPAD + q) * CDIM + h * HDIM + d] = (bf16)(U[nt][r] / rs);
      }
    }
  }
}

// ---------------------------------------------------------------------------
// LayerNorm: one block per (b,q) row of 1024 channels.
// ---------------------------------------------------------------------------
__global__ __launch_bounds__(256) void ln_kernel(
    const float* __restrict__ preln, const float* __restrict__ gamma,
    const float* __restrict__ beta, float* __restrict__ out) {
  __shared__ float red[2][8];
  const int b = blockIdx.x / QLEN, q = blockIdx.x % QLEN;
  const float* x = preln + ((long)b * QPAD + q) * CDIM;
  float s = 0.f, s2 = 0.f;
  for (int c = threadIdx.x; c < CDIM; c += 256) {
    const float v = x[c];
    s += v;
    s2 += v * v;
  }
  const int wv = threadIdx.x >> 5, lane = threadIdx.x & 31;
#pragma unroll
  for (int o = 16; o > 0; o >>= 1) {
    s += __shfl_xor(s, o);
    s2 += __shfl_xor(s2, o);
  }
  if (lane == 0) { red[0][wv] = s; red[1][wv] = s2; }
  __syncthreads();
  if (threadIdx.x == 0) {
    float a = 0.f, c2 = 0.f;
    for (int i = 0; i < 8; i++) { a += red[0][i]; c2 += red[1][i]; }
    red[0][0] = a;
    red[1][0] = c2;
  }
  __syncthreads();
  const float mean = red[0][0] * (1.f / CDIM);
  const float var = red[1][0] * (1.f / CDIM) - mean * mean;
  const float inv = rsqrtf(var + 1e-5f);
  float* o = out + ((long)b * QLEN + q) * CDIM;
  for (int c = threadIdx.x; c < CDIM; c += 256)
    o[c] = (x[c] - mean) * inv * gamma[c] + beta[c];
}

// ---------------------------------------------------------------------------
extern "C" void kernel_launch(void* const* d_in, const int* in_sizes, int n_in,
                              void* d_out, int out_size, void* d_ws,
                              size_t ws_size, hipStream_t stream) {
  (void)in_sizes; (void)n_in; (void)out_size; (void)ws_size;
  const float* tgt  = (const float*)d_in[0];
  const float* mem  = (const float*)d_in[1];
  const float* mpos = (const float*)d_in[2];
  const float* qpos = (const float*)d_in[3];
  const float* Wq   = (const float*)d_in[4];
  const float* Wk   = (const float*)d_in[5];
  const float* Wv   = (const float*)d_in[6];
  const float* Wo   = (const float*)d_in[7];
  const float* bo   = (const float*)d_in[8];
  const float* gam  = (const float*)d_in[9];
  const float* bet  = (const float*)d_in[10];
  float* out = (float*)d_out;

  char* ws = (char*)d_ws;
  bf16* Kp = (bf16*)ws;  ws += (size_t)BATCH * NHEAD * KEYS * HDIM * 2;  // 64 MB
  bf16* Vt = (bf16*)ws;  ws += (size_t)BATCH * NHEAD * KEYS * HDIM * 2;  // 64 MB
  bf16* Qp = (bf16*)ws;  ws += (size_t)BATCH * NHEAD * QPAD * HDIM * 2;
  bf16* upd = (bf16*)ws; ws += (size_t)BATCH * QPAD * CDIM * 2;
  float* preln = (float*)ws;  // BATCH*QPAD*CDIM*4

  const dim3 blk(256);
  // K/V projections: M = 8*4096 = 32768 rows -> 256 M-blocks x 16 N-groups.
  gemm_kernel<0><<<dim3(256, 16), blk, 0, stream>>>(mem, mpos, nullptr, Wk,
                                                    nullptr, nullptr, Kp, nullptr);
  gemm_kernel<1><<<dim3(256, 16), blk, 0, stream>>>(mem, nullptr, nullptr, Wv,
                                                    nullptr, nullptr, Vt, nullptr);
  // Q projection: M = 8*112 = 896 rows -> 7 M-blocks x 16 N-groups.
  gemm_kernel<2><<<dim3(7, 16), blk, 0, stream>>>(tgt, qpos, nullptr, Wq,
                                                  nullptr, nullptr, Qp, nullptr);
  // Attention: one block per (b, head).
  const size_t sm = (size_t)QPAD * 132 * 4 + (size_t)QPAD * 128 * 2 + QPAD * 4 +
                    (size_t)64 * VSTRIDE * 2;
  attn_kernel<<<BATCH * NHEAD, blk, sm, stream>>>(Qp, Kp, Vt, upd);
  // Output projection + bias + residual.
  gemm_kernel<3><<<dim3(7, 16), blk, 0, stream>>>(nullptr, nullptr, upd, Wo, bo,
                                                  tgt, nullptr, preln);
  // LayerNorm.
  ln_kernel<<<BATCH * QLEN, blk, 0, stream>>>(preln, gam, bet, out);
}
